// AttentionCell_11802570130233
// MI455X (gfx1250) — matrix operand order repaired and checked by reading, not compiled, not run on hardware
//
#include <hip/hip_runtime.h>

// ---------------------------------------------------------------------------
// AttentionCell for MI455X (gfx1250, wave32, WMMA).
//   scores = inputs @ questions^T / sqrt(D)   [N,N]
//   P      = softmax(scores, axis=-1)         [N,N]  (also an output)
//   out    = P @ inputs                       [N,D]
// d_out = [ out (N*D f32) | P (N*N f32) ]
//
// GEMMs run on v_wmma_f32_16x16x32_bf16 with:
//   - B tile staged in LDS (shared by the 8 waves of a block)
//   - ping-pong LDS buffers + 2x-unrolled, branch-free steady-state loop
//   - 2 M-tiles per wave: each LDS B-fragment feeds 2 WMMAs (8 wmma : 8 ds_load)
//   - global_prefetch_b8 on the streaming operands
// All GEMM operands are bf16 in workspace so they are L2-resident
// (bf16 P = 128 MiB < 192 MiB L2; f32 P would not fit).
// ---------------------------------------------------------------------------

typedef __bf16 bf16_t;
typedef __attribute__((ext_vector_type(16))) __bf16 v16bf;
typedef __attribute__((ext_vector_type(8)))  float  v8f;
typedef __attribute__((ext_vector_type(4)))  float  v4f;
typedef __attribute__((ext_vector_type(4)))  unsigned int v4u;
typedef __attribute__((ext_vector_type(2)))  unsigned int v2u;

static constexpr int kN = 8192;
static constexpr int kD = 1024;

// ---------------------------------------------------------------------------
// Kernel 1: f32 -> bf16 convert of inputs and questions (coalesced, 4-wide).
// ---------------------------------------------------------------------------
__global__ void convert_bf16_kernel(const float* __restrict__ a,
                                    const float* __restrict__ b,
                                    bf16_t* __restrict__ ab,
                                    bf16_t* __restrict__ bb)
{
    const size_t i4 = (size_t)blockIdx.x * blockDim.x + threadIdx.x;
    const size_t i  = i4 * 4;
    v4f va = *(const v4f*)(a + i);
    v4f vb = *(const v4f*)(b + i);
    union { bf16_t h[4]; v2u u; } pa, pb;
#pragma unroll
    for (int c = 0; c < 4; ++c) { pa.h[c] = (bf16_t)va[c]; pb.h[c] = (bf16_t)vb[c]; }
    *(v2u*)(ab + i) = pa.u;
    *(v2u*)(bb + i) = pb.u;
}

// ---------------------------------------------------------------------------
// Kernel 2: LDS-tiled transpose + convert: dst[d*N + n] = bf16(src[n*D + d]).
// Block (32,8), 32x32 tile.
// ---------------------------------------------------------------------------
__global__ void transpose_convert_kernel(const float* __restrict__ src,  // [rows, cols]
                                         bf16_t* __restrict__ dst,       // [cols, rows]
                                         int rows, int cols)
{
    __shared__ float tile[32][33];
    const int tx = threadIdx.x;
    const int ty = threadIdx.y;
    const int x  = blockIdx.x * 32 + tx;
    const int y0 = blockIdx.y * 32;
#pragma unroll
    for (int j = 0; j < 4; ++j)
        tile[ty + j * 8][tx] = src[(size_t)(y0 + ty + j * 8) * cols + x];
    __syncthreads();
    const int x2 = blockIdx.y * 32 + tx;
    const int y2 = blockIdx.x * 32;
#pragma unroll
    for (int j = 0; j < 4; ++j)
        dst[(size_t)(y2 + ty + j * 8) * rows + x2] = (bf16_t)tile[tx][ty + j * 8];
}

// ---------------------------------------------------------------------------
// Kernel 3/5: bf16 WMMA GEMM.  C[M,N] = scale * A[M,K] * B[K,N]
//   A   : row-major bf16, leading dim lda
//   Bcc : "column-contiguous" B, i.e. Bcc[n*ldb + k] == B[k][n] (row-major N x K)
//   C   : row-major f32, leading dim ldc
// Requires K % 64 == 0 and M % 256 == 0 (true here: K in {1024, 8192}, M = 8192).
//
// Block = 256 threads = 8 waves; block tile 256M x 64N.
// Wave tile 32M x 64N = 8 accumulators; K step 32, unrolled x2 over ping-pong
// LDS buffers. Per phase: 4 B fragments from LDS feed 2 A fragments -> 8 WMMAs.
//
// Fragment layouts per CDNA5 ISA 7.12.2 (wave32):
//   A 16x32 bf16 : lane l<16 -> row l,  K {0..7},{16..23}; lane>=16 -> {8..15},{24..31}
//   B 32x16 bf16 : lane l<16 -> col l, K {0..7},{16..23}; lane>=16 -> col l-16, K {8..15},{24..31}
//   C/D 16x16 f32: reg r, lane l<16 -> (M=r, N=l) ; lane>=16 -> (M=r+8, N=l-16)
// ---------------------------------------------------------------------------
__global__ __launch_bounds__(256)
void gemm_bf16_wmma_kernel(const bf16_t* __restrict__ A,
                           const bf16_t* __restrict__ Bcc,
                           float* __restrict__ C,
                           int K, int lda, int ldb, int ldc,
                           float scale)
{
    __shared__ __align__(16) bf16_t sB[2][64 * 40];  // 2 x (64 rows x 32 k, pad->40)

    const int t    = threadIdx.x;
    const int lane = t & 31;
    const int wave = t >> 5;
    const int l15  = lane & 15;
    const int half = lane >> 4;                      // 0 or 1

    const int m0 = blockIdx.y * 256 + wave * 32;     // 8 waves * 32 rows
    const int n0 = blockIdx.x * 64;                  // 4 tiles * 16 cols

    const bf16_t* arow0 = A + (size_t)(m0 + l15) * lda + half * 8;
    const bf16_t* arow1 = arow0 + (size_t)16 * lda;

    // B staging: thread t fetches 8 bf16 of row (n0 + t/4), k-chunk (t%4)*8.
    const int srow = t >> 2;
    const int scol = (t & 3) * 8;
    const bf16_t* bstage_g = Bcc + (size_t)(n0 + srow) * ldb + scol;
    bf16_t* sdst0 = &sB[0][srow * 40 + scol];
    bf16_t* sdst1 = &sB[1][srow * 40 + scol];

    const int lds_off = half * 8;

    union Frag { v16bf v; v4u u[2]; };

    v8f acc[2][4] = {};
    Frag a0[2], a1[2];
    v4u stage;

    // Helper: load the two A fragments (rows m0..m0+15 and m0+16..m0+31) at k.
    auto load_a = [&](Frag (&a)[2], int k) {
        a[0].u[0] = *(const v4u*)(arow0 + k);
        a[0].u[1] = *(const v4u*)(arow0 + k + 16);
        a[1].u[0] = *(const v4u*)(arow1 + k);
        a[1].u[1] = *(const v4u*)(arow1 + k + 16);
    };
    // Helper: 8 WMMAs against LDS buffer `buf`.
    auto mma_phase = [&](int buf, Frag (&a)[2]) {
#pragma unroll
        for (int j = 0; j < 4; ++j) {
            Frag b;
            const bf16_t* bp = &sB[buf][(j * 16 + l15) * 40 + lds_off];
            b.u[0] = *(const v4u*)(bp);
            b.u[1] = *(const v4u*)(bp + 16);
            acc[0][j] = __builtin_amdgcn_wmma_f32_16x16x32_bf16(
                false, a[0].v, false, b.v, (short)0, acc[0][j], false, false);
            acc[1][j] = __builtin_amdgcn_wmma_f32_16x16x32_bf16(
                false, a[1].v, false, b.v, (short)0, acc[1][j], false, false);
        }
    };

    // --- prologue: stage k = 0 into buf0 / a0 ---
    load_a(a0, 0);
    stage = *(const v4u*)(bstage_g);
    *(v4u*)sdst0 = stage;
    __syncthreads();

    int k0 = 0;
    for (; k0 + 64 < K; k0 += 64) {
        // ---- phase 1: compute k0 from buf0/a0, fetch k0+32 ----
        load_a(a1, k0 + 32);
        stage = *(const v4u*)(bstage_g + k0 + 32);
        __builtin_prefetch(arow0 + k0 + 256, 0, 3);      // global_prefetch_b8
        __builtin_prefetch(arow1 + k0 + 256, 0, 3);
        __builtin_prefetch(bstage_g + k0 + 256, 0, 3);
        mma_phase(0, a0);
        *(v4u*)sdst1 = stage;
        __syncthreads();

        // ---- phase 2: compute k0+32 from buf1/a1, fetch k0+64 ----
        load_a(a0, k0 + 64);
        stage = *(const v4u*)(bstage_g + k0 + 64);
        mma_phase(1, a1);
        *(v4u*)sdst0 = stage;
        __syncthreads();
    }

    // --- epilogue: last 64 of K (buf0/a0 already hold k0) ---
    load_a(a1, k0 + 32);
    stage = *(const v4u*)(bstage_g + k0 + 32);
    mma_phase(0, a0);
    *(v4u*)sdst1 = stage;
    __syncthreads();
    mma_phase(1, a1);

    // --- store: 2 M-tiles x 4 N-tiles ---
#pragma unroll
    for (int i = 0; i < 2; ++i)
#pragma unroll
        for (int j = 0; j < 4; ++j) {
            const int col = n0 + j * 16 + l15;
#pragma unroll
            for (int r = 0; r < 8; ++r) {
                const int row = m0 + i * 16 + half * 8 + r;
                C[(size_t)row * ldc + col] = acc[i][j][r] * scale;
            }
        }
}

// ---------------------------------------------------------------------------
// Kernel 4: row softmax over 8192 cols, block(256) per row.
// Normalizes P in place (f32) and writes bf16 copy for the second GEMM.
// ---------------------------------------------------------------------------
__global__ __launch_bounds__(256)
void softmax_row_kernel(float* __restrict__ P,        // [kN, kN] in/out
                        bf16_t* __restrict__ Pb)      // [kN, kN] bf16 out
{
    __shared__ float red[256];
    const int row = blockIdx.x;
    const int t   = threadIdx.x;
    float* prow = P + (size_t)row * kN;

    v4f v[8];
    float m = -3.402823466e38f;
#pragma unroll
    for (int i = 0; i < 8; ++i) {
        v[i] = *(const v4f*)(prow + t * 4 + i * 1024);
#pragma unroll
        for (int c = 0; c < 4; ++c) m = fmaxf(m, v[i][c]);
    }
    red[t] = m;
    __syncthreads();
#pragma unroll
    for (int s = 128; s > 0; s >>= 1) {
        if (t < s) red[t] = fmaxf(red[t], red[t + s]);
        __syncthreads();
    }
    m = red[0];
    __syncthreads();

    float sum = 0.f;
#pragma unroll
    for (int i = 0; i < 8; ++i)
#pragma unroll
        for (int c = 0; c < 4; ++c) {
            float e = __expf(v[i][c] - m);
            v[i][c] = e;
            sum += e;
        }
    red[t] = sum;
    __syncthreads();
#pragma unroll
    for (int s = 128; s > 0; s >>= 1) {
        if (t < s) red[t] += red[t + s];
        __syncthreads();
    }
    const float inv = 1.0f / red[0];

    bf16_t* brow = Pb + (size_t)row * kN;
#pragma unroll
    for (int i = 0; i < 8; ++i) {
        v[i] *= inv;
        *(v4f*)(prow + t * 4 + i * 1024) = v[i];
        union { bf16_t h[4]; v2u u; } pk;
#pragma unroll
        for (int c = 0; c < 4; ++c) pk.h[c] = (bf16_t)v[i][c];
        *(v2u*)(brow + t * 4 + i * 1024) = pk.u;
    }
}

// ---------------------------------------------------------------------------
// Launch
// ---------------------------------------------------------------------------
extern "C" void kernel_launch(void* const* d_in, const int* in_sizes, int n_in,
                              void* d_out, int out_size, void* d_ws, size_t ws_size,
                              hipStream_t stream)
{
    (void)in_sizes; (void)n_in; (void)out_size; (void)ws_size;

    const float* inputs    = (const float*)d_in[0];
    const float* questions = (const float*)d_in[1];

    float* out    = (float*)d_out;                       // [kN, kD]
    float* p_attn = out + (size_t)kN * kD;               // [kN, kN]

    const size_t nd = (size_t)kN * kD;                   // 8,388,608
    bf16_t* inputs_b    = (bf16_t*)d_ws;                 // 16 MiB
    bf16_t* questions_b = inputs_b + nd;                 // 16 MiB
    bf16_t* inputsT_b   = questions_b + nd;              // 16 MiB  [kD, kN]
    bf16_t* pattn_b     = inputsT_b + nd;                // 128 MiB [kN, kN]

    const float scale = 0.03125f;                        // 1/sqrt(1024)

    // 1) bf16 conversions (coalesced)
    {
        const int threads = 256;
        const int blocks  = (int)(nd / 4 / threads);     // 8192 blocks
        convert_bf16_kernel<<<blocks, threads, 0, stream>>>(
            inputs, questions, inputs_b, questions_b);
    }
    // 2) inputs^T bf16 [kD, kN]
    {
        dim3 grid(kD / 32, kN / 32);                     // (32, 256)
        dim3 block(32, 8);
        transpose_convert_kernel<<<grid, block, 0, stream>>>(inputs, inputsT_b, kN, kD);
    }
    // 3) scores = scale * inputs @ questions^T  -> p_attn region (f32)
    {
        dim3 grid(kN / 64, kN / 256);                    // (128, 32)
        gemm_bf16_wmma_kernel<<<grid, 256, 0, stream>>>(
            inputs_b, questions_b, p_attn,
            /*K=*/kD, /*lda=*/kD, /*ldb=*/kD, /*ldc=*/kN, scale);
    }
    // 4) softmax rows: normalize f32 in place + bf16 copy
    softmax_row_kernel<<<kN, 256, 0, stream>>>(p_attn, pattn_b);

    // 5) out = P @ inputs   (A = P_bf16 [kN,kN], Bcc = inputs^T bf16 [kD,kN])
    {
        dim3 grid(kD / 64, kN / 256);                    // (16, 32)
        gemm_bf16_wmma_kernel<<<grid, 256, 0, stream>>>(
            pattn_b, inputsT_b, out,
            /*K=*/kN, /*lda=*/kN, /*ldb=*/kN, /*ldc=*/kD, /*scale=*/1.0f);
    }
}
